// MambaEncoderLayer_120259084377
// MI455X (gfx1250) — matrix operand order repaired and checked by reading, not compiled, not run on hardware
//
#include <hip/hip_runtime.h>

typedef __attribute__((ext_vector_type(16))) _Float16 v16h;
typedef __attribute__((ext_vector_type(8)))  _Float16 v8h;
typedef __attribute__((ext_vector_type(8)))  float    v8f;

#define BB   2
#define LL   2048
#define DD   512
#define DIc  1024
#define DSc  16
#define DRc  32
#define MM   (BB * LL)     /* 4096 token rows */
#define NXP  64            /* DR + 2*DS */

// ---------------------------------------------------------------------------
// f32 -> f16 converters (operand prep for WMMA)
// ---------------------------------------------------------------------------
__global__ void cvt_f32_f16(const float* __restrict__ s, _Float16* __restrict__ d, int n) {
  int i = blockIdx.x * blockDim.x + threadIdx.x;
  if (i < n) d[i] = (_Float16)s[i];
}

__global__ void cvt_f32_f16_strided(const float* __restrict__ s, int sld,
                                    _Float16* __restrict__ d, int dld,
                                    int rows, int cols) {
  int i = blockIdx.x * blockDim.x + threadIdx.x;
  if (i >= rows * cols) return;
  int r = i / cols, c = i - r * cols;
  d[(size_t)r * dld + c] = (_Float16)s[(size_t)r * sld + c];
}

// ---------------------------------------------------------------------------
// Register-blocked NT GEMM: C[m,n] = sum_k A[m,k] * W[n,k]
// (A:[M,K] f16 row-major, W:[N,K] f16 row-major, C:[M,N] f32)
// One wave32 computes a 32x64 C block = 2 M-tiles x 4 N-tiles.
// Per 32-wide k-step: 2 A-tile loads + 4 B-tile loads -> 8 WMMAs.
// Prefetch is unconditional (speculative prefetch never faults; reading
// <=64B past the operand buffer is harmless) to avoid a branch that forced
// an address spill + full loadcnt wait in the previous build.
// Requires M % 32 == 0, N % 64 == 0, K % 32 == 0 (true for all four GEMMs).
// ---------------------------------------------------------------------------
__global__ void gemm_nt_wmma(const _Float16* __restrict__ A, int lda,
                             const _Float16* __restrict__ W, int ldw,
                             float* __restrict__ C, int ldc,
                             int M, int N, int K) {
  const int lane = threadIdx.x & 31;
  const int wave = blockIdx.x * (blockDim.x >> 5) + (threadIdx.x >> 5);
  const int bN = N >> 6;                 // 64-wide N blocks
  const int bM = M >> 5;                 // 32-tall M blocks
  if (wave >= bM * bN) return;
  const int tM = (wave / bN) << 5;
  const int tN = (wave % bN) << 6;

  const int l15   = lane & 15;
  const int koffA = (lane >> 4) << 4;    // 0 or 16
  const int koffB = (lane >> 4) << 4;    // 0 or 16

  const _Float16* aRow[2];
  aRow[0] = A + (size_t)(tM + l15) * lda;
  aRow[1] = A + (size_t)(tM + 16 + l15) * lda;
  const _Float16* wRow[4];
#pragma unroll
  for (int j = 0; j < 4; ++j)
    wRow[j] = W + (size_t)(tN + 16 * j + l15) * ldw + koffB;

  v8f acc[2][4] = {};
  for (int k0 = 0; k0 < K; k0 += 32) {
    union { v16h v; v8h h[2]; } aU[2];
#pragma unroll
    for (int i = 0; i < 2; ++i) {
      aU[i].h[0] = *reinterpret_cast<const v8h*>(aRow[i] + k0 + koffA);
      aU[i].h[1] = *reinterpret_cast<const v8h*>(aRow[i] + k0 + 8 + koffA);
    }
    v16h bt[4];
#pragma unroll
    for (int j = 0; j < 4; ++j)
      bt[j] = *reinterpret_cast<const v16h*>(wRow[j] + k0);
    __builtin_prefetch(aRow[0] + k0 + 32, 0, 0);
    __builtin_prefetch(wRow[0] + k0 + 32, 0, 0);
#pragma unroll
    for (int i = 0; i < 2; ++i)
#pragma unroll
      for (int j = 0; j < 4; ++j)
        acc[i][j] = __builtin_amdgcn_wmma_f32_16x16x32_f16(
            false, aU[i].v, false, bt[j], (short)0, acc[i][j], false, false);
  }

#pragma unroll
  for (int i = 0; i < 2; ++i) {
    const int mbase = tM + 16 * i + ((lane >> 4) << 3);
#pragma unroll
    for (int j = 0; j < 4; ++j) {
      const int ncol = tN + 16 * j + l15;
#pragma unroll
      for (int v = 0; v < 8; ++v)
        C[(size_t)(mbase + v) * ldc + ncol] = acc[i][j][v];
    }
  }
}

// ---------------------------------------------------------------------------
// Depthwise causal conv1d (DC=4) + SiLU over the xi half of xz.
// Writes u (f32, needed by scan) and uh (f16, A-operand of x_proj GEMM).
// ---------------------------------------------------------------------------
__global__ void conv_silu_kernel(const float* __restrict__ xz,
                                 const float* __restrict__ cw,
                                 const float* __restrict__ cb,
                                 float* __restrict__ u,
                                 _Float16* __restrict__ uh) {
  int idx = blockIdx.x * blockDim.x + threadIdx.x;
  if (idx >= MM * DIc) return;
  int d  = idx & (DIc - 1);
  int ml = idx >> 10;           // row index m = b*L + l
  int l  = ml & (LL - 1);
  float acc = cb[d];
#pragma unroll
  for (int j = 0; j < 4; ++j) {
    int lt = l - 3 + j;
    if (lt >= 0)
      acc = fmaf(cw[d * 4 + j], xz[(size_t)(ml - 3 + j) * (2 * DIc) + d], acc);
  }
  float sv = acc / (1.f + __expf(-acc));   // SiLU
  u[idx]  = sv;
  uh[idx] = (_Float16)sv;
}

// ---------------------------------------------------------------------------
// Selective scan. lane = (channel, state): 2 channels per wave, s = lane&15.
// Loop-carried dep is one FMA; y reduced over the 16 states with shfl_xor.
// s==0 lane fuses +u*D, SiLU(z) gate, and writes y as f16 for out_proj.
// ---------------------------------------------------------------------------
__global__ void scan_kernel(const float* __restrict__ xz,
                            const float* __restrict__ u,
                            const float* __restrict__ xdbl,
                            const float* __restrict__ dpre,
                            const float* __restrict__ b_dt,
                            const float* __restrict__ A_log,
                            const float* __restrict__ D_par,
                            _Float16* __restrict__ yh) {
  int wave = blockIdx.x * (blockDim.x >> 5) + (threadIdx.x >> 5);
  int lane = threadIdx.x & 31;
  int s  = lane & 15;
  int ch = (wave << 1) + (lane >> 4);
  if (ch >= BB * DIc) return;
  int b = ch >> 10;             // ch / DI
  int d = ch & (DIc - 1);

  float Av  = -__expf(A_log[d * DSc + s]);
  float bdt = b_dt[d];
  float Dp  = D_par[d];
  float h   = 0.f;
  size_t mbase = (size_t)b * LL;

  for (int t = 0; t < LL; ++t) {
    size_t m  = mbase + t;
    float dp  = dpre[m * DIc + d] + bdt;
    float delta = (dp > 20.f) ? dp : __logf(1.f + __expf(dp));   // softplus
    float uv  = u[m * DIc + d];
    float Bv  = xdbl[m * NXP + DRc + s];
    float Cv  = xdbl[m * NXP + DRc + DSc + s];
    float dA  = __expf(delta * Av);
    h = fmaf(dA, h, delta * Bv * uv);       // the only carried dependency
    float p = h * Cv;
    p += __shfl_xor(p, 1, 32);
    p += __shfl_xor(p, 2, 32);
    p += __shfl_xor(p, 4, 32);
    p += __shfl_xor(p, 8, 32);
    if (s == 0) {
      float zv = xz[m * (2 * DIc) + DIc + d];
      float g  = zv / (1.f + __expf(-zv));  // SiLU gate
      yh[m * DIc + d] = (_Float16)((p + uv * Dp) * g);
    }
  }
}

// ---------------------------------------------------------------------------
// LayerNorm(eps=1e-6) + residual. One wave per 512-wide row.
// ---------------------------------------------------------------------------
__global__ void ln_res_kernel(const float* __restrict__ om,
                              const float* __restrict__ x,
                              const float* __restrict__ ln_w,
                              const float* __restrict__ ln_b,
                              float* __restrict__ out) {
  int wave = blockIdx.x * (blockDim.x >> 5) + (threadIdx.x >> 5);
  int lane = threadIdx.x & 31;
  if (wave >= MM) return;
  const float* row = om + (size_t)wave * DD;
  float vals[16];
  float sum = 0.f, sq = 0.f;
#pragma unroll
  for (int i = 0; i < 16; ++i) {
    float v = row[lane + 32 * i];
    vals[i] = v; sum += v; sq = fmaf(v, v, sq);
  }
#pragma unroll
  for (int msk = 1; msk < 32; msk <<= 1) {
    sum += __shfl_xor(sum, msk, 32);
    sq  += __shfl_xor(sq,  msk, 32);
  }
  float mu   = sum * (1.f / DD);
  float var  = sq * (1.f / DD) - mu * mu;
  float rstd = rsqrtf(var + 1e-6f);
#pragma unroll
  for (int i = 0; i < 16; ++i) {
    int c = lane + 32 * i;
    out[(size_t)wave * DD + c] =
        x[(size_t)wave * DD + c] + (vals[i] - mu) * rstd * ln_w[c] + ln_b[c];
  }
}

// ---------------------------------------------------------------------------
extern "C" void kernel_launch(void* const* d_in, const int* in_sizes, int n_in,
                              void* d_out, int out_size, void* d_ws, size_t ws_size,
                              hipStream_t stream) {
  const float* x      = (const float*)d_in[0];
  const float* W_in   = (const float*)d_in[1];
  const float* conv_w = (const float*)d_in[2];
  const float* conv_b = (const float*)d_in[3];
  const float* W_xprj = (const float*)d_in[4];
  const float* W_dt   = (const float*)d_in[5];
  const float* b_dt   = (const float*)d_in[6];
  const float* A_log  = (const float*)d_in[7];
  const float* D_par  = (const float*)d_in[8];
  const float* W_out  = (const float*)d_in[9];
  const float* ln_w   = (const float*)d_in[10];
  const float* ln_b   = (const float*)d_in[11];
  float* out = (float*)d_out;
  char*  ws  = (char*)d_ws;

  size_t off = 0;
  auto alloc = [&](size_t bytes) {
    size_t o = off; off += (bytes + 255) & ~(size_t)255; return o;
  };
  _Float16* xh     = (_Float16*)(ws + alloc((size_t)MM * DD * 2));
  _Float16* Winh   = (_Float16*)(ws + alloc((size_t)2 * DIc * DD * 2));
  _Float16* Wxph   = (_Float16*)(ws + alloc((size_t)NXP * DIc * 2));
  _Float16* Wdth   = (_Float16*)(ws + alloc((size_t)DIc * DRc * 2));
  _Float16* Wouth  = (_Float16*)(ws + alloc((size_t)DD * DIc * 2));
  float*    xz     = (float*)   (ws + alloc((size_t)MM * 2 * DIc * 4));
  float*    u_     = (float*)   (ws + alloc((size_t)MM * DIc * 4));
  _Float16* uh     = (_Float16*)(ws + alloc((size_t)MM * DIc * 2));
  float*    xdbl   = (float*)   (ws + alloc((size_t)MM * NXP * 4));
  _Float16* dth    = (_Float16*)(ws + alloc((size_t)MM * DRc * 2));
  float*    dpre   = (float*)   (ws + alloc((size_t)MM * DIc * 4));
  _Float16* yh     = (_Float16*)(ws + alloc((size_t)MM * DIc * 2));
  float*    om     = (float*)   (ws + alloc((size_t)MM * DD * 4));
  (void)ws_size; (void)in_sizes; (void)n_in; (void)out_size;

  auto cdiv = [](long a, long b) { return (int)((a + b - 1) / b); };

  // Operand prep (f32 -> f16)
  cvt_f32_f16<<<cdiv((long)MM * DD, 256), 256, 0, stream>>>(x, xh, MM * DD);
  cvt_f32_f16<<<cdiv((long)2 * DIc * DD, 256), 256, 0, stream>>>(W_in, Winh, 2 * DIc * DD);
  cvt_f32_f16<<<cdiv((long)NXP * DIc, 256), 256, 0, stream>>>(W_xprj, Wxph, NXP * DIc);
  cvt_f32_f16<<<cdiv((long)DIc * DRc, 256), 256, 0, stream>>>(W_dt, Wdth, DIc * DRc);
  cvt_f32_f16<<<cdiv((long)DD * DIc, 256), 256, 0, stream>>>(W_out, Wouth, DD * DIc);

  // G1: xz = x @ W_in^T   [4096,512]x[2048,512] -> [4096,2048]
  {
    int tiles = (MM / 32) * ((2 * DIc) / 64);
    gemm_nt_wmma<<<cdiv(tiles, 8), 256, 0, stream>>>(xh, DD, Winh, DD,
                                                     xz, 2 * DIc, MM, 2 * DIc, DD);
  }

  // Depthwise conv + SiLU -> u (f32 + f16)
  conv_silu_kernel<<<cdiv((long)MM * DIc, 256), 256, 0, stream>>>(xz, conv_w, conv_b, u_, uh);

  // G2: x_dbl = u @ W_xproj^T   [4096,1024]x[64,1024] -> [4096,64]
  {
    int tiles = (MM / 32) * (NXP / 64);
    gemm_nt_wmma<<<cdiv(tiles, 8), 256, 0, stream>>>(uh, DIc, Wxph, DIc,
                                                     xdbl, NXP, MM, NXP, DIc);
  }

  // dt slice (cols 0..31 of x_dbl) -> f16 compact
  cvt_f32_f16_strided<<<cdiv((long)MM * DRc, 256), 256, 0, stream>>>(
      xdbl, NXP, dth, DRc, MM, DRc);

  // G3: delta_pre = dt @ W_dt^T   [4096,32]x[1024,32] -> [4096,1024]
  {
    int tiles = (MM / 32) * (DIc / 64);
    gemm_nt_wmma<<<cdiv(tiles, 8), 256, 0, stream>>>(dth, DRc, Wdth, DRc,
                                                     dpre, DIc, MM, DIc, DRc);
  }

  // Selective scan (+ gate) -> yh (f16)
  scan_kernel<<<(BB * DIc / 2) / 8, 256, 0, stream>>>(xz, u_, xdbl, dpre,
                                                      b_dt, A_log, D_par, yh);

  // G4: out_m = y @ W_out^T   [4096,1024]x[512,1024] -> [4096,512]
  {
    int tiles = (MM / 32) * (DD / 64);
    gemm_nt_wmma<<<cdiv(tiles, 8), 256, 0, stream>>>(yh, DIc, Wouth, DIc,
                                                     om, DD, MM, DD, DIc);
  }

  // LayerNorm + residual -> out
  ln_res_kernel<<<MM / 8, 256, 0, stream>>>(om, x, ln_w, ln_b, out);
}
